// GlobalMoEModel_60163901883070
// MI455X (gfx1250) — matrix operand order repaired and checked by reading, not compiled, not run on hardware
//
#include <hip/hip_runtime.h>
#include <hip/hip_bf16.h>
#include <stdint.h>

#define DEVINL __device__ __forceinline__

typedef _Float16 f16;
typedef __attribute__((ext_vector_type(16))) _Float16 v16h;
typedef __attribute__((ext_vector_type(8)))  _Float16 v8h;
typedef __attribute__((ext_vector_type(8)))  float    v8f;

constexpr int cV  = 32000;   // vocab
constexpr int cH  = 768;     // hidden
constexpr int cL  = 4;       // layers
constexpr int cNH = 12;      // attn heads
constexpr int cNKV= 4;       // kv heads
constexpr int cHD = 64;      // head dim
constexpr int cE  = 64;      // experts
constexpr int cF  = 192;     // moe intermediate
constexpr int cK  = 8;       // top-k
constexpr int cT  = 512;     // tokens (B=1)
constexpr float cEPS = 1e-6f;

union Frag { v16h v; v8h h[2]; };

DEVINL v8f wmma16(v16h a, v16h b, v8f c) {
  return __builtin_amdgcn_wmma_f32_16x16x32_f16(false, a, false, b, (short)0, c,
                                                false, false);
}

// CDNA5 async copy global -> LDS (16B per lane), tracked by ASYNCcnt.
DEVINL void async_ld_b128(unsigned lds_off, const void* gptr) {
  unsigned long long ga = (unsigned long long)(uintptr_t)gptr;
  asm volatile("global_load_async_to_lds_b128 %0, %1, off"
               :: "v"(lds_off), "v"(ga) : "memory");
}
DEVINL void wait_async_le4() { asm volatile("s_wait_asynccnt 4" ::: "memory"); }
DEVINL void wait_async0()    { asm volatile("s_wait_asynccnt 0" ::: "memory"); }

// ---------------------------------------------------------------------------
// f16 WMMA GEMM, all operands in [row][K-contiguous] layout:
//   A[M,K] (lda), Bt[N,K] (ldb)  ->  C = A * Bt^T
// Batched over blockIdx.z: A += z*sAz, Bt += (z/bdiv)*sBz, C/C16 += z*s{C,C16}z.
// Outputs: optional f32 C; optional f16 C16 (c16t=1 stores transposed [N][M]).
// Block = 128 threads (4 waves as 2x2), macro tile 64x64, wave tile 32x32.
// Double-buffered LDS panels filled with global_load_async_to_lds_b128.
// M,N multiples of 64; K multiple of 32.
// ---------------------------------------------------------------------------
__global__ __launch_bounds__(128)
void k_gemm16(const f16* __restrict__ A, int lda, long long sAz,
              const f16* __restrict__ Bt, int ldb, long long sBz, int bdiv,
              float* __restrict__ C, int ldc, long long sCz,
              f16* __restrict__ C16, int ldc16, long long sC16z, int c16t,
              int K) {
  constexpr int LS = 40;                 // f16 row stride in LDS (80B)
  __shared__ f16 sA[2][64 * LS];
  __shared__ f16 sB[2][64 * LS];
  const int z = blockIdx.z;
  A  += (size_t)z * sAz;
  Bt += (size_t)(z / bdiv) * sBz;
  const int tid  = threadIdx.x;
  const int lane = tid & 31, wave = tid >> 5;
  const int rr = lane & 15, hi = lane >> 4;
  const int wm = wave >> 1, wn = wave & 1;
  const int m0 = blockIdx.x * 64, n0 = blockIdx.y * 64;

  auto stage = [&](int buf, int k0) {
    unsigned aoff = (unsigned)(uintptr_t)&sA[buf][0];
    unsigned boff = (unsigned)(uintptr_t)&sB[buf][0];
#pragma unroll
    for (int c = tid; c < 256; c += 128) {     // 2 iterations, 4 async/wave total
      int row = c >> 2, part = c & 3;
      async_ld_b128(aoff + row * (LS * 2) + part * 16,
                    A + (size_t)(m0 + row) * lda + k0 + part * 8);
      async_ld_b128(boff + row * (LS * 2) + part * 16,
                    Bt + (size_t)(n0 + row) * ldb + k0 + part * 8);
    }
  };

  v8f acc[2][2] = {};
  stage(0, 0);
  for (int k0 = 0; k0 < K; k0 += 32) {
    const int buf = (k0 >> 5) & 1;
    const bool more = (k0 + 32) < K;
    if (more) { stage(buf ^ 1, k0 + 32); wait_async_le4(); }
    else      { wait_async0(); }
    __syncthreads();

    Frag a[2], b[2];
#pragma unroll
    for (int i = 0; i < 2; ++i) {
      int arow = wm * 32 + i * 16 + rr;
      a[i].h[0] = *(const v8h*)&sA[buf][arow * LS + hi * 8];
      a[i].h[1] = *(const v8h*)&sA[buf][arow * LS + 16 + hi * 8];
      int bcol = wn * 32 + i * 16 + rr;
      b[i].h[0] = *(const v8h*)&sB[buf][bcol * LS + hi * 8];
      b[i].h[1] = *(const v8h*)&sB[buf][bcol * LS + 16 + hi * 8];
    }
#pragma unroll
    for (int i = 0; i < 2; ++i)
#pragma unroll
      for (int j2 = 0; j2 < 2; ++j2)
        acc[i][j2] = wmma16(a[i].v, b[j2].v, acc[i][j2]);
    __syncthreads();
  }

  if (C)   C   += (size_t)z * sCz;
  if (C16) C16 += (size_t)z * sC16z;
#pragma unroll
  for (int i = 0; i < 2; ++i) {
#pragma unroll
    for (int j2 = 0; j2 < 2; ++j2) {
      int colg = n0 + wn * 32 + j2 * 16 + rr;
#pragma unroll
      for (int j = 0; j < 8; ++j) {
        int rowg = m0 + wm * 32 + i * 16 + hi * 8 + j;
        float v = acc[i][j2][j];
        if (C) C[(size_t)rowg * ldc + colg] = v;
        if (C16) {
          if (c16t) C16[(size_t)colg * ldc16 + rowg] = (f16)v;
          else      C16[(size_t)rowg * ldc16 + colg] = (f16)v;
        }
      }
    }
  }
}

// ---------------------------------------------------------------------------
// f16 fragment helpers for the gathered expert GEMMs (contiguous b128 loads)
// ---------------------------------------------------------------------------
DEVINL v16h load_a_g16(const f16* __restrict__ row, int k0, int hi) {
  Frag f;
  f.h[0] = *(const v8h*)(row + k0 + hi * 8);
  f.h[1] = *(const v8h*)(row + k0 + 16 + hi * 8);
  return f.v;
}
// Bt is [N][K] (K contiguous)
DEVINL v16h load_bt_g16(const f16* __restrict__ Bt, int ldk, int k0, int col, int hi) {
  Frag f;
  const f16* p = Bt + (size_t)col * ldk + k0 + hi * 8;
  f.h[0] = *(const v8h*)p;
  f.h[1] = *(const v8h*)(p + 16);
  return f.v;
}

// ---------------------------------------------------------------------------
// Elementwise / small kernels
// ---------------------------------------------------------------------------
__global__ void k_cvt(const float* __restrict__ s, f16* __restrict__ d, int n) {
  int i = blockIdx.x * blockDim.x + threadIdx.x;
  if (i < n) d[i] = (f16)s[i];
}

// Batched f32[K][N] -> f16[N][K] transpose-convert (batch = blockIdx.z).
__global__ void k_cvt_t(const float* __restrict__ s, f16* __restrict__ d,
                        int Kd, int Nd) {
  const float* sb = s + (size_t)blockIdx.z * Kd * Nd;
  f16* db = d + (size_t)blockIdx.z * Kd * Nd;
  int i = blockIdx.x * blockDim.x + threadIdx.x;
  if (i < Kd * Nd) {
    int k = i / Nd, n = i % Nd;
    db[(size_t)n * Kd + k] = (f16)sb[i];
  }
}

__global__ void k_embed(const float* __restrict__ embed, const int* __restrict__ ids,
                        float* __restrict__ x) {
  int t = blockIdx.x;
  int id = ids[t];
  for (int i = threadIdx.x; i < cH; i += blockDim.x)
    x[(size_t)t * cH + i] = embed[(size_t)id * cH + i];
}

__global__ void k_rmsnorm(const float* __restrict__ x, const float* __restrict__ w,
                          f16* __restrict__ y) {
  int t = blockIdx.x;
  __shared__ float red[256];
  int tid = threadIdx.x;
  float s = 0.f;
  for (int i = tid; i < cH; i += 256) { float v = x[(size_t)t * cH + i]; s += v * v; }
  red[tid] = s; __syncthreads();
  for (int o = 128; o > 0; o >>= 1) { if (tid < o) red[tid] += red[tid + o]; __syncthreads(); }
  float inv = rsqrtf(red[0] / (float)cH + cEPS);
  for (int i = tid; i < cH; i += 256)
    y[(size_t)t * cH + i] = (f16)(x[(size_t)t * cH + i] * inv * w[i]);
}

__global__ void k_add(float* __restrict__ x, const float* __restrict__ y, int n) {
  int i = blockIdx.x * blockDim.x + threadIdx.x;
  if (i < n) x[i] += y[i];
}

__global__ void k_zero_f(float* p, int n) {
  int i = blockIdx.x * blockDim.x + threadIdx.x;
  if (i < n) p[i] = 0.f;
}

__global__ void k_zero_i(int* p, int n) {
  int i = blockIdx.x * blockDim.x + threadIdx.x;
  if (i < n) p[i] = 0;
}

// RoPE: read f32 q/k, write rotated f16 q16/k16.
__global__ void k_rope(const float* __restrict__ q, const float* __restrict__ k,
                       f16* __restrict__ q16, f16* __restrict__ k16) {
  int t = blockIdx.x;
  for (int idx = threadIdx.x; idx < (cNH + cNKV) * (cHD / 2); idx += blockDim.x) {
    int head = idx / (cHD / 2), d = idx % (cHD / 2);
    float inv_freq = __powf(10000.f, -(float)(2 * d) / (float)cHD);
    float ang = (float)t * inv_freq;
    float c = __cosf(ang), s = __sinf(ang);
    bool isq = head < cNH;
    const float* p = isq ? q + (size_t)t * (cNH * cHD) + head * cHD
                         : k + (size_t)t * (cNKV * cHD) + (head - cNH) * cHD;
    f16* o = isq ? q16 + (size_t)t * (cNH * cHD) + head * cHD
                 : k16 + (size_t)t * (cNKV * cHD) + (head - cNH) * cHD;
    float x0 = p[d], x1 = p[d + cHD / 2];
    o[d] = (f16)(x0 * c - x1 * s);
    o[d + cHD / 2] = (f16)(x1 * c + x0 * s);
  }
}

// Causal softmax over scores [NH,T,T] with 1/sqrt(HD); writes f16 probs.
__global__ void k_softmax(float* __restrict__ scores, f16* __restrict__ p16) {
  int tq = blockIdx.x, h = blockIdx.y, tid = threadIdx.x;
  float* row = scores + ((size_t)h * cT + tq) * cT;
  f16* orow = p16 + ((size_t)h * cT + tq) * cT;
  const float scale = 0.125f;
  __shared__ float red[256];
  float mx = -3.0e38f;
  for (int j = tid; j < cT; j += 256) {
    float v = (j <= tq) ? row[j] * scale : -3.0e38f;
    mx = fmaxf(mx, v);
  }
  red[tid] = mx; __syncthreads();
  for (int o = 128; o > 0; o >>= 1) { if (tid < o) red[tid] = fmaxf(red[tid], red[tid + o]); __syncthreads(); }
  mx = red[0]; __syncthreads();
  float s = 0.f;
  for (int j = tid; j < cT; j += 256) {
    float v = (j <= tq) ? __expf(row[j] * scale - mx) : 0.f;
    row[j] = v; s += v;
  }
  red[tid] = s; __syncthreads();
  for (int o = 128; o > 0; o >>= 1) { if (tid < o) red[tid] += red[tid + o]; __syncthreads(); }
  float inv = 1.f / red[0];
  for (int j = tid; j < cT; j += 256) orow[j] = (f16)(row[j] * inv);
}

// Router: softmax over E, top-8 (renormalized), per-expert token lists,
// aux-loss sums accumulated across all layers.
__global__ void k_router(const float* __restrict__ rl, int* __restrict__ counts,
                         int* __restrict__ tok_list, float* __restrict__ wgt_list,
                         float* __restrict__ selcount, float* __restrict__ probsum) {
  int t = blockIdx.x * blockDim.x + threadIdx.x;
  if (t >= cT) return;
  float p[cE];
  float mx = -3.0e38f;
  for (int e = 0; e < cE; ++e) { p[e] = rl[(size_t)t * cE + e]; mx = fmaxf(mx, p[e]); }
  float s = 0.f;
  for (int e = 0; e < cE; ++e) { p[e] = __expf(p[e] - mx); s += p[e]; }
  float inv = 1.f / s;
  for (int e = 0; e < cE; ++e) { p[e] *= inv; atomicAdd(&probsum[e], p[e]); }
  float pc[cE];
  for (int e = 0; e < cE; ++e) pc[e] = p[e];
  float tp[cK]; int tix[cK]; float tsum = 0.f;
  for (int kk = 0; kk < cK; ++kk) {
    int bi = 0; float bv = -1.f;
    for (int e = 0; e < cE; ++e) if (pc[e] > bv) { bv = pc[e]; bi = e; }
    tp[kk] = bv; tix[kk] = bi; pc[bi] = -2.f; tsum += bv;
  }
  float winv = 1.f / tsum;
  for (int kk = 0; kk < cK; ++kk) {
    int e = tix[kk];
    atomicAdd(&selcount[e], 1.f);
    int pos = atomicAdd(&counts[e], 1);
    tok_list[e * cT + pos] = t;
    wgt_list[e * cT + pos] = tp[kk] * winv;
  }
}

// Gathered expert up+gate GEMM, fused silu(g)*u*route_weight, f16 out.
// Wg/Wu supplied transposed: [E][F][H] (K=H contiguous).
__global__ __launch_bounds__(32)
void k_expert_up(const f16* __restrict__ normed,
                 const f16* __restrict__ Wgt, const f16* __restrict__ Wut,
                 const int* __restrict__ counts, const int* __restrict__ tok_list,
                 const float* __restrict__ wgt_list, f16* __restrict__ act) {
  const int e = blockIdx.x;
  const int cnt = counts[e];
  const int ti = blockIdx.y / (cF / 16);
  const int fj = blockIdx.y % (cF / 16);
  if (ti * 16 >= cnt) return;
  const int lane = threadIdx.x & 31, rr = lane & 15, hi = lane >> 4;
  int slot = ti * 16 + rr;
  int sl = slot < cnt ? slot : cnt - 1;
  const int tk = tok_list[e * cT + sl];
  const f16* Arow = normed + (size_t)tk * cH;
  const f16* Bg = Wgt + (size_t)e * cF * cH;
  const f16* Bu = Wut + (size_t)e * cF * cH;
  const int n0 = fj * 16;
  v8f g = {}, u = {};
  for (int k0 = 0; k0 < cH; k0 += 32) {
    v16h a = load_a_g16(Arow, k0, hi);
    v16h bg = load_bt_g16(Bg, cH, k0, n0 + rr, hi);
    v16h bu = load_bt_g16(Bu, cH, k0, n0 + rr, hi);
    g = wmma16(a, bg, g);
    u = wmma16(a, bu, u);
  }
  const int col = n0 + rr;
#pragma unroll
  for (int j = 0; j < 8; ++j) {
    int s = ti * 16 + hi * 8 + j;
    if (s < cnt) {
      float w = wgt_list[e * cT + s];
      float gv = g[j];
      float sv = gv / (1.f + __expf(-gv));  // silu
      act[((size_t)e * cT + s) * cF + col] = (f16)(sv * u[j] * w);
    }
  }
}

// Gathered expert down GEMM, scatter-accumulate into residual stream x.
// Wd supplied transposed: [E][H][F] (K=F contiguous).
__global__ __launch_bounds__(32)
void k_expert_down(const f16* __restrict__ act, const f16* __restrict__ Wdt,
                   const int* __restrict__ counts, const int* __restrict__ tok_list,
                   float* __restrict__ x) {
  const int e = blockIdx.x;
  const int cnt = counts[e];
  const int ti = blockIdx.y / (cH / 16);
  const int hj = blockIdx.y % (cH / 16);
  if (ti * 16 >= cnt) return;
  const int lane = threadIdx.x & 31, rr = lane & 15, hi = lane >> 4;
  const f16* Arow = act + ((size_t)e * cT + ti * 16 + rr) * cF;
  const f16* B = Wdt + (size_t)e * cH * cF;
  const int n0 = hj * 16;
  v8f acc = {};
  for (int k0 = 0; k0 < cF; k0 += 32) {
    v16h a = load_a_g16(Arow, k0, hi);
    v16h b = load_bt_g16(B, cF, k0, n0 + rr, hi);
    acc = wmma16(a, b, acc);
  }
  const int col = n0 + rr;
#pragma unroll
  for (int j = 0; j < 8; ++j) {
    int s = ti * 16 + hi * 8 + j;
    if (s < cnt) {
      int tk = tok_list[e * cT + s];
      atomicAdd(&x[(size_t)tk * cH + col], acc[j]);
    }
  }
}

// Cross entropy per token over V logits; accumulates (sum nll, valid count).
__global__ void k_ce(const float* __restrict__ logits, const int* __restrict__ labels,
                     float* __restrict__ ce_acc) {
  int t = blockIdx.x, tid = threadIdx.x;
  const float* row = logits + (size_t)t * cV;
  __shared__ float red[256];
  float mx = -3.0e38f;
  for (int j = tid; j < cV; j += 256) mx = fmaxf(mx, row[j]);
  red[tid] = mx; __syncthreads();
  for (int o = 128; o > 0; o >>= 1) { if (tid < o) red[tid] = fmaxf(red[tid], red[tid + o]); __syncthreads(); }
  mx = red[0]; __syncthreads();
  float s = 0.f;
  for (int j = tid; j < cV; j += 256) s += __expf(row[j] - mx);
  red[tid] = s; __syncthreads();
  for (int o = 128; o > 0; o >>= 1) { if (tid < o) red[tid] += red[tid + o]; __syncthreads(); }
  if (tid == 0) {
    int lbl = labels[t];
    if (lbl != -100) {
      int cl = lbl < 0 ? 0 : lbl;
      float nll = (mx + __logf(red[0])) - row[cl];
      atomicAdd(&ce_acc[0], nll);
      atomicAdd(&ce_acc[1], 1.f);
    }
  }
}

__global__ void k_final(const float* __restrict__ ce_acc,
                        const float* __restrict__ selcount,
                        const float* __restrict__ probsum,
                        float* __restrict__ loss_out) {
  if (threadIdx.x == 0 && blockIdx.x == 0) {
    float aux = 0.f;
    for (int e = 0; e < cE; ++e) aux += selcount[e] * probsum[e];
    float R = (float)(cL * cT);
    aux = aux * (float)cE / (R * R);
    float ce = ce_acc[0] / fmaxf(ce_acc[1], 1.f);
    loss_out[0] = ce + 0.01f * aux;
  }
}

// ---------------------------------------------------------------------------
// Host orchestration
// ---------------------------------------------------------------------------
extern "C" void kernel_launch(void* const* d_in, const int* in_sizes, int n_in,
                              void* d_out, int out_size, void* d_ws, size_t ws_size,
                              hipStream_t stream) {
  const int*   input_ids = (const int*)d_in[0];
  const int*   labels    = (const int*)d_in[1];
  const float* embed     = (const float*)d_in[2];
  const float* ln1       = (const float*)d_in[3];
  const float* Wq        = (const float*)d_in[4];
  const float* Wk        = (const float*)d_in[5];
  const float* Wv        = (const float*)d_in[6];
  const float* Wo        = (const float*)d_in[7];
  const float* ln2       = (const float*)d_in[8];
  const float* router_w  = (const float*)d_in[9];
  const float* Wg        = (const float*)d_in[10];
  const float* Wu        = (const float*)d_in[11];
  const float* Wd        = (const float*)d_in[12];
  const float* fnw       = (const float*)d_in[13];

  float* logits = (float*)d_out;                 // [T, V] f32
  float* loss   = logits + (size_t)cT * cV;

  char* base = (char*)d_ws;
  auto alloc = [&](size_t bytes) {
    void* p = (void*)base;
    base += (bytes + 255) & ~(size_t)255;
    return p;
  };
  // f16 weight mirrors, pre-transposed to [N][K] where noted
  f16* embed16 = (f16*)alloc((size_t)cV * cH * 2);                  // [V][H] (already NK)
  f16* Wq16t   = (f16*)alloc((size_t)cL * cH * cNH * cHD * 2);      // [L][768][768]
  f16* Wk16t   = (f16*)alloc((size_t)cL * cH * cNKV * cHD * 2);     // [L][256][768]
  f16* Wv16t   = (f16*)alloc((size_t)cL * cH * cNKV * cHD * 2);     // [L][256][768]
  f16* Wo16t   = (f16*)alloc((size_t)cL * cNH * cHD * cH * 2);      // [L][768][768]
  f16* rw16t   = (f16*)alloc((size_t)cL * cH * cE * 2);             // [L][64][768]
  f16* Wg16t   = (f16*)alloc((size_t)cE * cH * cF * 2);             // [E][192][768]
  f16* Wu16t   = (f16*)alloc((size_t)cE * cH * cF * 2);             // [E][192][768]
  f16* Wd16t   = (f16*)alloc((size_t)cE * cF * cH * 2);             // [E][768][192]
  // activations
  float* x       = (float*)alloc((size_t)cT * cH * 4);
  f16*   normed16= (f16*)alloc((size_t)cT * cH * 2);
  float* qb      = (float*)alloc((size_t)cT * cNH * cHD * 4);
  float* kb      = (float*)alloc((size_t)cT * cNKV * cHD * 4);
  f16*   q16     = (f16*)alloc((size_t)cT * cNH * cHD * 2);
  f16*   k16     = (f16*)alloc((size_t)cT * cNKV * cHD * 2);
  f16*   v16t    = (f16*)alloc((size_t)cNKV * cHD * cT * 2);        // [256][512] transposed
  float* scores  = (float*)alloc((size_t)cNH * cT * cT * 4);
  f16*   p16     = (f16*)alloc((size_t)cNH * cT * cT * 2);
  f16*   attno16 = (f16*)alloc((size_t)cT * cNH * cHD * 2);
  float* attnp   = (float*)alloc((size_t)cT * cH * 4);
  float* rl_all  = (float*)alloc((size_t)cL * cT * cE * 4);
  f16*   act16   = (f16*)alloc((size_t)cE * cT * cF * 2);
  float* wgtlist = (float*)alloc((size_t)cE * cT * 4);
  float* selcnt  = (float*)alloc(cE * 4);
  float* probsum = (float*)alloc(cE * 4);
  float* ce_acc  = (float*)alloc(2 * 4);
  int*   counts  = (int*)alloc(cE * 4);
  int*   toklist = (int*)alloc((size_t)cE * cT * 4);

  auto cvt = [&](const float* s, f16* d, size_t n) {
    k_cvt<<<dim3((unsigned)((n + 255) / 256)), 256, 0, stream>>>(s, d, (int)n);
  };
  auto cvt_t = [&](const float* s, f16* d, int Kd, int Nd, int batch) {
    dim3 g((Kd * Nd + 255) / 256, 1, batch);
    k_cvt_t<<<g, 256, 0, stream>>>(s, d, Kd, Nd);
  };
  auto gemm = [&](const f16* A, int lda, long long sAz,
                  const f16* Bt, int ldb, long long sBz, int bdiv,
                  float* C, int ldc, long long sCz,
                  f16* C16, int ldc16, long long sC16z, int c16t,
                  int M, int N, int K, int nz) {
    dim3 g(M / 64, N / 64, nz);
    k_gemm16<<<g, 128, 0, stream>>>(A, lda, sAz, Bt, ldb, sBz, bdiv,
                                    C, ldc, sCz, C16, ldc16, sC16z, c16t, K);
  };

  // ---- one-time (per launch) weight conversion/transposition to f16 ----
  cvt(embed, embed16, (size_t)cV * cH);
  cvt_t(Wq,       Wq16t, cH, cNH * cHD,  cL);
  cvt_t(Wk,       Wk16t, cH, cNKV * cHD, cL);
  cvt_t(Wv,       Wv16t, cH, cNKV * cHD, cL);
  cvt_t(Wo,       Wo16t, cNH * cHD, cH,  cL);
  cvt_t(router_w, rw16t, cH, cE,         cL);
  cvt_t(Wg,       Wg16t, cH, cF,         cE);
  cvt_t(Wu,       Wu16t, cH, cF,         cE);
  cvt_t(Wd,       Wd16t, cF, cH,         cE);

  // init accumulators
  k_zero_f<<<1, 64, 0, stream>>>(selcnt, cE);
  k_zero_f<<<1, 64, 0, stream>>>(probsum, cE);
  k_zero_f<<<1, 64, 0, stream>>>(ce_acc, 2);

  // embedding gather
  k_embed<<<cT, 256, 0, stream>>>(embed, input_ids, x);

  for (int i = 0; i < cL; ++i) {
    // ---- attention ----
    k_rmsnorm<<<cT, 256, 0, stream>>>(x, ln1 + (size_t)i * cH, normed16);
    gemm(normed16, cH, 0, Wq16t + (size_t)i * cH * cNH * cHD, cH, 0, 1,
         qb, cNH * cHD, 0, nullptr, 0, 0, 0, cT, cNH * cHD, cH, 1);
    gemm(normed16, cH, 0, Wk16t + (size_t)i * cH * cNKV * cHD, cH, 0, 1,
         kb, cNKV * cHD, 0, nullptr, 0, 0, 0, cT, cNKV * cHD, cH, 1);
    gemm(normed16, cH, 0, Wv16t + (size_t)i * cH * cNKV * cHD, cH, 0, 1,
         nullptr, 0, 0, v16t, cT, 0, 1 /*transposed store*/, cT, cNKV * cHD, cH, 1);
    k_rope<<<cT, 256, 0, stream>>>(qb, kb, q16, k16);
    // scores: batched over heads (z=NH), GQA via bdiv=3
    gemm(q16, cNH * cHD, cHD, k16, cNKV * cHD, cHD, cNH / cNKV,
         scores, cT, (long long)cT * cT, nullptr, 0, 0, 0, cT, cT, cHD, cNH);
    k_softmax<<<dim3(cT, cNH), 256, 0, stream>>>(scores, p16);
    // O = P @ V: batched over heads, B = v16t [256][512]
    gemm(p16, cT, (long long)cT * cT, v16t, cT, (long long)cHD * cT, cNH / cNKV,
         nullptr, 0, 0, attno16, cNH * cHD, cHD, 0, cT, cHD, cT, cNH);
    gemm(attno16, cNH * cHD, 0, Wo16t + (size_t)i * cNH * cHD * cH, cNH * cHD, 0, 1,
         attnp, cH, 0, nullptr, 0, 0, 0, cT, cH, cNH * cHD, 1);
    k_add<<<(cT * cH + 255) / 256, 256, 0, stream>>>(x, attnp, cT * cH);

    // ---- MoE ----
    k_rmsnorm<<<cT, 256, 0, stream>>>(x, ln2 + (size_t)i * cH, normed16);
    float* rl = rl_all + (size_t)i * cT * cE;
    gemm(normed16, cH, 0, rw16t + (size_t)i * cH * cE, cH, 0, 1,
         rl, cE, 0, nullptr, 0, 0, 0, cT, cE, cH, 1);
    k_zero_i<<<1, 64, 0, stream>>>(counts, cE);
    k_router<<<(cT + 255) / 256, 256, 0, stream>>>(rl, counts, toklist, wgtlist,
                                                   selcnt, probsum);
    k_expert_up<<<dim3(cE, (cT / 16) * (cF / 16)), 32, 0, stream>>>(
        normed16, Wg16t, Wu16t, counts, toklist, wgtlist, act16);
    k_expert_down<<<dim3(cE, (cT / 16) * (cH / 16)), 32, 0, stream>>>(
        act16, Wd16t, counts, toklist, x);
  }

  // ---- head ----
  k_rmsnorm<<<cT, 256, 0, stream>>>(x, fnw, normed16);
  gemm(normed16, cH, 0, embed16, cH, 0, 1,
       logits, cV, 0, nullptr, 0, 0, 0, cT, cV, cH, 1);
  k_ce<<<cT, 256, 0, stream>>>(logits, labels, ce_acc);
  k_final<<<1, 32, 0, stream>>>(ce_acc, selcnt, probsum, loss);

  (void)in_sizes; (void)n_in; (void)out_size; (void)ws_size;
}
